// LinearAttention_20976620273899
// MI455X (gfx1250) — compile-verified, matrix operand order checked by reading, not run on hardware
//
#include <hip/hip_runtime.h>
#include <hip/hip_bf16.h>
#include <math.h>

// ---------------------------------------------------------------------------
// LinearAttention on MI455X (gfx1250), bf16 WMMA pipeline with ctx->Wout fusion
// Shapes: B=4, N=8192, D=1024, H=16, DH=64, INNER=1024, 3*INNER=3072
//   K0a: x (f32) -> xb (bf16)                                 [32768,1024]
//   K0b: Wqkv -> Wt_qkv bf16 transposed  [3072,1024] (row = out col, K contig)
//   K0c: Wout -> Wt_out  bf16 transposed [1024,1024]
//   K1 : qkvb = xb @ Wqkv  (WMMA bf16, 2x2 register blocking) [32768,3072]
//   K2 : q <- softmax(q*scale) over dh (in place, bf16)
//   K3 : k <- exp(k)/colsum  (3 deterministic passes, no atomics)
//   K5 : ctx[b,h] = q^T @ k  (TDM tile->LDS + LDS transpose, split-K + reduce)
//   K4 : Weff_t[b] = (blockdiag(ctx) @ Wout)^T  bf16          [4,1024,1024]
//   K6 : out = v @ Weff[b] + bout  (WMMA bf16 -> f32, 2x2 blocking)
// ---------------------------------------------------------------------------

typedef __attribute__((ext_vector_type(16))) __bf16 v16bf;
typedef __attribute__((ext_vector_type(8)))  float  v8f;
typedef __attribute__((ext_vector_type(4)))  unsigned u32x4;
typedef __attribute__((ext_vector_type(8)))  int      i32x8;
typedef __attribute__((ext_vector_type(4)))  int      i32x4;

union Frag { v16bf v; uint4 q[2]; unsigned short s[16]; };

__device__ __forceinline__ unsigned short f2bf(float f) {
  union { float f; unsigned u; } c; c.f = f;
  unsigned u = c.u;
  return (unsigned short)((u + 0x7fffu + ((u >> 16) & 1u)) >> 16);  // RNE
}
__device__ __forceinline__ float bf2f(unsigned short h) {
  union { unsigned u; float f; } c; c.u = ((unsigned)h) << 16;
  return c.f;
}

// --------- Tensor Data Mover: 2-D bf16 tile (global, strided) -> LDS --------
// D# group0: count=1 | lds_addr | global_addr[56:0] | type=2
// D# group1: data_size=1(2B); tensor_dim0/1 = tile_dim0/1; dim0 stride given.
// 6-arg builtin (clang-23/therock): (g0, g1, g2, g3, g_extra, cpol);
// groups 2/3 + extra zero-filled (2-D tensor, no iterate/gather).
__device__ __forceinline__ void tdm_load_tile_2d(unsigned lds_off,
                                                 const void* gptr,
                                                 unsigned tile_w,   // elems, contig
                                                 unsigned tile_h,   // rows
                                                 unsigned stride) { // elems per row
  unsigned long long ga = (unsigned long long)(size_t)gptr;
  u32x4 g0;
  g0[0] = 1u;                                      // count=1 valid descriptor
  g0[1] = lds_off;                                 // LDS byte address
  g0[2] = (unsigned)(ga & 0xffffffffull);          // global_addr[31:0]
  g0[3] = (unsigned)((ga >> 32) & 0x1ffffffull)    // global_addr[56:32]
        | (2u << 30);                              // type=2 ("image")
  i32x8 g1;
  g1[0] = (int)(1u << 16);                         // data_size=1 -> 2 bytes
  g1[1] = (int)((tile_w & 0xffffu) << 16);         // tensor_dim0[15:0]
  g1[2] = (int)(((tile_w >> 16) & 0xffffu) | ((tile_h & 0xffffu) << 16));
  g1[3] = (int)(((tile_h >> 16) & 0xffffu) | ((tile_w & 0xffffu) << 16));
  g1[4] = (int)(tile_h & 0xffffu);                 // tile_dim1 (tile_dim2=0)
  g1[5] = (int)stride;                             // tensor_dim0_stride[31:0]
  g1[6] = 0;
  g1[7] = 0;
  i32x4 z4 = {0, 0, 0, 0};
  i32x8 z8 = {0, 0, 0, 0, 0, 0, 0, 0};
  __builtin_amdgcn_tensor_load_to_lds(g0, g1, z4, z4, z8, 0);
}

// ------------------------------ K0: conversions -----------------------------

__global__ __launch_bounds__(256) void k_cvt_x(const float* __restrict__ x,
                                               unsigned short* __restrict__ xb) {
  int i = blockIdx.x * 256 + threadIdx.x;           // one float4 per thread
  float4 v = ((const float4*)x)[i];
  union { unsigned short s[4]; uint2 u; } p;
  p.s[0] = f2bf(v.x); p.s[1] = f2bf(v.y); p.s[2] = f2bf(v.z); p.s[3] = f2bf(v.w);
  ((uint2*)xb)[i] = p.u;
}

__global__ __launch_bounds__(256) void k_tr_wqkv(const float* __restrict__ w,
                                                 unsigned short* __restrict__ wt) {
  int i = blockIdx.x * 256 + threadIdx.x;           // 1024*3072 elements
  int n = i % 3072, k = i / 3072;
  wt[(size_t)n * 1024 + k] = f2bf(w[(size_t)k * 3072 + n]);
}

__global__ __launch_bounds__(256) void k_tr_wout(const float* __restrict__ w,
                                                 unsigned short* __restrict__ wt) {
  int i = blockIdx.x * 256 + threadIdx.x;           // 1024*1024 elements
  int j = i % 1024, d = i / 1024;
  wt[(size_t)j * 1024 + d] = f2bf(w[(size_t)d * 1024 + j]);
}

// ------------------------------ K1: qkv GEMM -------------------------------
// C[32768,3072] = A @ B^T; 128x128 block tile, 16 waves, 32x32 per wave (2x2)
__global__ __launch_bounds__(512) void k_gemm_qkv(const unsigned short* __restrict__ xb,
                                                  const unsigned short* __restrict__ wt,
                                                  unsigned short* __restrict__ qkvb) {
  const int w    = threadIdx.x >> 5;
  const int lane = threadIdx.x & 31;
  const int lr = lane & 15, half = lane >> 4;
  const int wr = w >> 2, wc = w & 3;
  const unsigned short* ar[2];
  const unsigned short* br[2];
  for (int i = 0; i < 2; ++i) {
    ar[i] = xb + (size_t)(blockIdx.y * 128 + wr * 32 + i * 16 + lr) * 1024;
    br[i] = wt + (size_t)(blockIdx.x * 128 + wc * 32 + i * 16 + lr) * 1024;
  }
  v8f acc[2][2] = {};
  for (int ks = 0; ks < 1024; ks += 32) {
    const int kf = ks + half * 8;
    __builtin_prefetch(ar[0] + kf + 64, 0, 3);
    __builtin_prefetch(br[0] + kf + 64, 0, 3);
    __builtin_prefetch(ar[1] + kf + 64, 0, 3);
    __builtin_prefetch(br[1] + kf + 64, 0, 3);
    Frag a[2], b[2];
    for (int i = 0; i < 2; ++i) {
      a[i].q[0] = *(const uint4*)(ar[i] + kf);
      a[i].q[1] = *(const uint4*)(ar[i] + kf + 16);
      b[i].q[0] = *(const uint4*)(br[i] + kf);
      b[i].q[1] = *(const uint4*)(br[i] + kf + 16);
    }
    for (int i = 0; i < 2; ++i)
      for (int j = 0; j < 2; ++j)
        acc[i][j] = __builtin_amdgcn_wmma_f32_16x16x32_bf16(
            false, a[i].v, false, b[j].v, (short)0, acc[i][j], false, false);
  }
  for (int i = 0; i < 2; ++i) {
    const int outm = blockIdx.y * 128 + wr * 32 + i * 16 + half * 8;  // M=r+8*half
    for (int j = 0; j < 2; ++j) {
      const int outn = blockIdx.x * 128 + wc * 32 + j * 16 + lr;      // N=lane%16
      for (int r = 0; r < 8; ++r)
        qkvb[(size_t)(outm + r) * 3072 + outn] = f2bf(acc[i][j][r]);
    }
  }
}

// --------------------- K2: q softmax over head dim (64) ---------------------
__global__ __launch_bounds__(256) void k_qsoft(unsigned short* __restrict__ qkvb) {
  int i = blockIdx.x * 256 + threadIdx.x;           // 32768*16 (row, head)
  int row = i >> 4, h = i & 15;
  unsigned short* base = qkvb + (size_t)row * 3072 + h * 64;
  union { uint4 u[8]; unsigned short s[64]; } V;
  for (int j = 0; j < 8; ++j) V.u[j] = ((const uint4*)base)[j];
  float m = -3.4e38f;
  for (int j = 0; j < 64; ++j) { float f = bf2f(V.s[j]) * 0.125f; m = fmaxf(m, f); }
  float sum = 0.f;
  for (int j = 0; j < 64; ++j) sum += expf(bf2f(V.s[j]) * 0.125f - m);
  float inv = 1.0f / sum;
  for (int j = 0; j < 64; ++j)
    V.s[j] = f2bf(expf(bf2f(V.s[j]) * 0.125f - m) * inv);
  for (int j = 0; j < 8; ++j) ((uint4*)base)[j] = V.u[j];
}

// ------------------- K3: k softmax over sequence dim (8192) -----------------
__global__ __launch_bounds__(256) void k_ksum_part(const unsigned short* __restrict__ qkvb,
                                                   float* __restrict__ psum) {
  const int col = blockIdx.x * 256 + threadIdx.x;   // 0..1023
  const int nc  = blockIdx.y;                       // 0..15
  const int b   = blockIdx.z;                       // 0..3
  float s = 0.f;
  for (int n = nc * 512; n < nc * 512 + 512; ++n)
    s += expf(bf2f(qkvb[(size_t)(b * 8192 + n) * 3072 + 1024 + col]));
  psum[(size_t)(b * 1024 + col) * 16 + nc] = s;
}

__global__ __launch_bounds__(256) void k_ksum_red(const float* __restrict__ psum,
                                                  float* __restrict__ kinv) {
  int i = blockIdx.x * 256 + threadIdx.x;           // 4096
  float s = 0.f;
  for (int j = 0; j < 16; ++j) s += psum[(size_t)i * 16 + j];
  kinv[i] = 1.0f / s;
}

__global__ __launch_bounds__(256) void k_knorm(unsigned short* __restrict__ qkvb,
                                               const float* __restrict__ kinv) {
  int i = blockIdx.x * 256 + threadIdx.x;           // 32768*1024
  int row = i >> 10, col = i & 1023, b = row >> 13;
  size_t a = (size_t)row * 3072 + 1024 + col;
  qkvb[a] = f2bf(expf(bf2f(qkvb[a])) * kinv[b * 1024 + col]);
}

// ------- K5: ctx partials: ctxp[b,h,nc] = q^T @ k over 1024-row chunk -------
// TDM pulls 32x64 bf16 tiles of q and k into LDS; threads transpose into
// WMMA fragment layout; one 16x16 tile per wave.
__global__ __launch_bounds__(512) void k_ctx(const unsigned short* __restrict__ qkvb,
                                             float* __restrict__ ctxp) {
  const int bh = blockIdx.x, b = bh >> 4, h = bh & 15;
  const int nc = blockIdx.y;                        // 0..7 K-chunk
  __shared__ unsigned short rawQ[32 * 64];          // [n][d] as in memory
  __shared__ unsigned short rawK[32 * 64];
  __shared__ unsigned short ldsA[64 * 32];          // [d][n], n contiguous
  __shared__ unsigned short ldsB[64 * 32];          // [e][n]
  const int t = threadIdx.x;
  const int w = t >> 5, lane = t & 31, lr = lane & 15, half = lane >> 4;
  const int tr = w >> 2, tc = w & 3;
  const int d4 = (t & 15) * 4;                      // 4 consecutive head dims
  const int nn = t >> 4;                            // 0..31 rows of chunk
  const size_t rowbase = (size_t)b * 8192 * 3072;
  const unsigned short* qbase = qkvb + rowbase + h * 64;
  const unsigned short* kbase = qkvb + rowbase + 1024 + h * 64;
  const unsigned ldsQoff = (unsigned)(size_t)rawQ;  // low 32 bits = LDS offset
  const unsigned ldsKoff = (unsigned)(size_t)rawK;
  v8f acc = {};
  for (int n0 = nc * 1024; n0 < nc * 1024 + 1024; n0 += 32) {
    if (w == 0) {                                   // one wave drives the TDM
      tdm_load_tile_2d(ldsQoff, qbase + (size_t)n0 * 3072, 64, 32, 3072);
      tdm_load_tile_2d(ldsKoff, kbase + (size_t)n0 * 3072, 64, 32, 3072);
      __builtin_amdgcn_s_wait_tensorcnt(0);
    }
    __syncthreads();                                // raw tiles ready
    union { uint2 u; unsigned short s[4]; } Q, K;
    Q.u = *(const uint2*)&rawQ[nn * 64 + d4];
    K.u = *(const uint2*)&rawK[nn * 64 + d4];
    for (int i = 0; i < 4; ++i) {
      ldsA[(d4 + i) * 32 + nn] = Q.s[i];            // transpose into [d][n]
      ldsB[(d4 + i) * 32 + nn] = K.s[i];
    }
    __syncthreads();                                // fragment layout ready
    Frag a, bb;
    const int nf = half * 8;
    a.q[0]  = *(const uint4*)&ldsA[(tr * 16 + lr) * 32 + nf];
    a.q[1]  = *(const uint4*)&ldsA[(tr * 16 + lr) * 32 + nf + 16];
    bb.q[0] = *(const uint4*)&ldsB[(tc * 16 + lr) * 32 + nf];
    bb.q[1] = *(const uint4*)&ldsB[(tc * 16 + lr) * 32 + nf + 16];
    acc = __builtin_amdgcn_wmma_f32_16x16x32_bf16(false, a.v, false, bb.v,
                                                  (short)0, acc, false, false);
  }
  float* cb = ctxp + ((size_t)bh * 8 + nc) * 4096;
  const int d0 = tr * 16 + half * 8, e = tc * 16 + lr;
  for (int r = 0; r < 8; ++r) cb[(d0 + r) * 64 + e] = acc[r];
}

__global__ __launch_bounds__(256) void k_ctx_red(const float* __restrict__ ctxp,
                                                 float* __restrict__ ctx) {
  int i = blockIdx.x * 256 + threadIdx.x;           // 64*4096
  int bh = i >> 12, off = i & 4095;
  float s = 0.f;
  for (int c = 0; c < 8; ++c) s += ctxp[((size_t)bh * 8 + c) * 4096 + off];
  ctx[(size_t)bh * 4096 + off] = s;
}

// ---------- K4: Weff_t[b][j][h*64+d] = sum_e ctx[b,h][d][e]*Wout[h*64+e][j] --
__global__ __launch_bounds__(512) void k_weff(const float* __restrict__ ctx,
                                              const unsigned short* __restrict__ wtout,
                                              unsigned short* __restrict__ wefft) {
  const int bh = blockIdx.y, b = bh >> 4, h = bh & 15;
  const int jbase = blockIdx.x * 64;
  const int w = threadIdx.x >> 5, lane = threadIdx.x & 31;
  const int lr = lane & 15, half = lane >> 4;
  const int tr = w >> 2, tc = w & 3;
  const float* ap = ctx + (size_t)bh * 4096 + (tr * 16 + lr) * 64;
  const int j = jbase + tc * 16 + lr;
  const unsigned short* bp = wtout + (size_t)j * 1024 + h * 64;
  v8f acc = {};
  for (int ks = 0; ks < 64; ks += 32) {
    const int kf = ks + half * 8;
    Frag a, bb;
    for (int i = 0; i < 8; ++i) a.s[i]     = f2bf(ap[kf + i]);
    for (int i = 0; i < 8; ++i) a.s[8 + i] = f2bf(ap[kf + 16 + i]);
    bb.q[0] = *(const uint4*)(bp + kf);
    bb.q[1] = *(const uint4*)(bp + kf + 16);
    acc = __builtin_amdgcn_wmma_f32_16x16x32_bf16(false, a.v, false, bb.v,
                                                  (short)0, acc, false, false);
  }
  union { unsigned short s[8]; uint4 q; } pk;
  for (int r = 0; r < 8; ++r) pk.s[r] = f2bf(acc[r]);   // d = tr*16+half*8+r
  *(uint4*)(wefft + (size_t)(b * 1024 + j) * 1024 + h * 64 + tr * 16 + half * 8) = pk.q;
}

// --------------------- K6: out = v @ Weff[b] + bout -------------------------
__global__ __launch_bounds__(512) void k_gemm_out(const unsigned short* __restrict__ qkvb,
                                                  const unsigned short* __restrict__ wefft,
                                                  const float* __restrict__ bout,
                                                  float* __restrict__ out) {
  const int w = threadIdx.x >> 5, lane = threadIdx.x & 31;
  const int lr = lane & 15, half = lane >> 4;
  const int wr = w >> 2, wc = w & 3;
  const int b = blockIdx.y >> 6;                    // 64 row-tiles(128) per batch
  const unsigned short* ar[2];
  const unsigned short* br[2];
  for (int i = 0; i < 2; ++i) {
    ar[i] = qkvb + (size_t)(blockIdx.y * 128 + wr * 32 + i * 16 + lr) * 3072 + 2048;
    br[i] = wefft + (size_t)(b * 1024 + blockIdx.x * 128 + wc * 32 + i * 16 + lr) * 1024;
  }
  v8f acc[2][2] = {};
  for (int ks = 0; ks < 1024; ks += 32) {
    const int kf = ks + half * 8;
    __builtin_prefetch(ar[0] + kf + 64, 0, 3);
    __builtin_prefetch(br[0] + kf + 64, 0, 3);
    __builtin_prefetch(ar[1] + kf + 64, 0, 3);
    __builtin_prefetch(br[1] + kf + 64, 0, 3);
    Frag a[2], bb[2];
    for (int i = 0; i < 2; ++i) {
      a[i].q[0]  = *(const uint4*)(ar[i] + kf);
      a[i].q[1]  = *(const uint4*)(ar[i] + kf + 16);
      bb[i].q[0] = *(const uint4*)(br[i] + kf);
      bb[i].q[1] = *(const uint4*)(br[i] + kf + 16);
    }
    for (int i = 0; i < 2; ++i)
      for (int j = 0; j < 2; ++j)
        acc[i][j] = __builtin_amdgcn_wmma_f32_16x16x32_bf16(
            false, a[i].v, false, bb[j].v, (short)0, acc[i][j], false, false);
  }
  for (int i = 0; i < 2; ++i) {
    const int outm = blockIdx.y * 128 + wr * 32 + i * 16 + half * 8;
    for (int j = 0; j < 2; ++j) {
      const int outn = blockIdx.x * 128 + wc * 32 + j * 16 + lr;
      const float bj = bout[outn];
      for (int r = 0; r < 8; ++r)
        out[(size_t)(outm + r) * 1024 + outn] = acc[i][j][r] + bj;
    }
  }
}

// ------------------------------- launcher -----------------------------------

extern "C" void kernel_launch(void* const* d_in, const int* in_sizes, int n_in,
                              void* d_out, int out_size, void* d_ws, size_t ws_size,
                              hipStream_t stream) {
  const float* x    = (const float*)d_in[0];   // [4,8192,1024]
  const float* Wqkv = (const float*)d_in[1];   // [1024,3072]
  const float* Wout = (const float*)d_in[2];   // [1024,1024]
  const float* bout = (const float*)d_in[3];   // [1024]
  float* out = (float*)d_out;                  // [32768,1024]

  char* ws = (char*)d_ws;
  unsigned short* xb     = (unsigned short*)(ws + 0);           //  64 MB
  unsigned short* wtqkv  = (unsigned short*)(ws + 67108864);    //   6 MB
  unsigned short* wtout  = (unsigned short*)(ws + 73400320);    //   2 MB
  unsigned short* qkvb   = (unsigned short*)(ws + 75497472);    // 192 MB
  float*          ctx    = (float*)        (ws + 276824064);    //   1 MB
  float*          kinv   = (float*)        (ws + 277872640);    //  16 KB
  float*          psum   = (float*)        (ws + 277889024);    // 256 KB
  unsigned short* wefft  = (unsigned short*)(ws + 278151168);   //   8 MB
  float*          ctxp   = (float*)        (ws + 286539776);    //   8 MB

  // K0: conversions / transposes
  k_cvt_x  <<<dim3(32768), dim3(256), 0, stream>>>(x, xb);
  k_tr_wqkv<<<dim3(12288), dim3(256), 0, stream>>>(Wqkv, wtqkv);
  k_tr_wout<<<dim3(4096),  dim3(256), 0, stream>>>(Wout, wtout);

  // K1: qkv = x @ Wqkv   (grid: 24 col-tiles x 256 row-tiles of 128x128)
  k_gemm_qkv<<<dim3(24, 256), dim3(512), 0, stream>>>(xb, wtqkv, qkvb);

  // K2: q softmax over head dim
  k_qsoft<<<dim3(2048), dim3(256), 0, stream>>>(qkvb);

  // K3: k softmax over sequence dim (deterministic 3-pass)
  k_ksum_part<<<dim3(4, 16, 4), dim3(256), 0, stream>>>(qkvb, psum);
  k_ksum_red <<<dim3(16),       dim3(256), 0, stream>>>(psum, kinv);
  k_knorm    <<<dim3(131072),   dim3(256), 0, stream>>>(qkvb, kinv);

  // K5: ctx[b,h] = q^T @ k_norm   (TDM-staged, split-K: 64 x 8, then reduce)
  k_ctx    <<<dim3(64, 8), dim3(512), 0, stream>>>(qkvb, ctxp);
  k_ctx_red<<<dim3(1024),  dim3(256), 0, stream>>>(ctxp, ctx);

  // K4: Weff_t[b] = (blockdiag(ctx) @ Wout)^T
  k_weff<<<dim3(16, 64), dim3(512), 0, stream>>>(ctx, wtout, wefft);

  // K6: out = v @ Weff[b] + bout  (grid: 8 col-tiles x 256 row-tiles)
  k_gemm_out<<<dim3(8, 256), dim3(512), 0, stream>>>(qkvb, wefft, bout, out);
}
